// CVRP_Decoder_38946763440472
// MI455X (gfx1250) — matrix operand order repaired and checked by reading, not compiled, not run on hardware
//
#include <hip/hip_runtime.h>

// ---------------- types ----------------
typedef __attribute__((ext_vector_type(16))) __bf16          v16bf;
typedef __attribute__((ext_vector_type(16))) unsigned short  v16u;
typedef __attribute__((ext_vector_type(8)))  unsigned short  v8u;
typedef __attribute__((ext_vector_type(8)))  float           v8f;

#define DEV __device__ __forceinline__

static constexpr int Bd = 32, Nn = 1000, Dd = 128, Ff = 512;
static constexpr int NP = 1024;          // padded K for AFT GEMM
static constexpr int BN = Bd * Nn;       // 32000 flat rows

DEV unsigned short f2bf(float f) {
  unsigned int u = __builtin_bit_cast(unsigned int, f);
  u += 0x7fffu + ((u >> 16) & 1u);       // round-to-nearest-even
  return (unsigned short)(u >> 16);
}

DEV v8f wmma_bf16(v16u a, v16u b, v8f c) {
  return __builtin_amdgcn_wmma_f32_16x16x32_bf16(
      false, __builtin_bit_cast(v16bf, a),
      false, __builtin_bit_cast(v16bf, b),
      (short)0, c, false, false);
}

// A fragment (16x32, M x K) from row-major fp32, converted to bf16.
// ISA 16-bit A layout: lane L -> row L%16; elems 0..7 = K(8h..8h+7),
// elems 8..15 = K(16+8h..16+8h+7), h = L/16.
DEV v16u loadA_f32(const float* base, int ld, int k0, int lane) {
  int r = lane & 15, h = (lane >> 4) & 1;
  const float* p = base + (size_t)r * ld + k0 + 8 * h;
  float4 x0 = reinterpret_cast<const float4*>(p)[0];
  float4 x1 = reinterpret_cast<const float4*>(p)[1];
  float4 x2 = reinterpret_cast<const float4*>(p + 16)[0];
  float4 x3 = reinterpret_cast<const float4*>(p + 16)[1];
  v16u u;
  u[0]=f2bf(x0.x); u[1]=f2bf(x0.y); u[2]=f2bf(x0.z); u[3]=f2bf(x0.w);
  u[4]=f2bf(x1.x); u[5]=f2bf(x1.y); u[6]=f2bf(x1.z); u[7]=f2bf(x1.w);
  u[8]=f2bf(x2.x); u[9]=f2bf(x2.y); u[10]=f2bf(x2.z); u[11]=f2bf(x2.w);
  u[12]=f2bf(x3.x); u[13]=f2bf(x3.y); u[14]=f2bf(x3.z); u[15]=f2bf(x3.w);
  return u;
}

// A fragment from row-major bf16 (global or LDS; two 16B contiguous loads).
DEV v16u loadA_bf(const unsigned short* base, int ld, int k0, int lane) {
  int r = lane & 15, h = (lane >> 4) & 1;
  const unsigned short* p = base + (size_t)r * ld + k0 + 8 * h;
  v8u a0 = *reinterpret_cast<const v8u*>(p);
  v8u a1 = *reinterpret_cast<const v8u*>(p + 16);
  return __builtin_shufflevector(a0, a1, 0,1,2,3,4,5,6,7,8,9,10,11,12,13,14,15);
}

// B fragment (32x16, K x N) from a transposed [Nout][K] bf16 array.
// ISA 16-bit B layout: lane n -> col n%16; elems 0..7 = K(8h..8h+7),
// elems 8..15 = K(16+8h..16+8h+7), h = n/16 (two 16B loads, same as A).
DEV v16u loadB_bf(const unsigned short* base, int ld, int n0, int k0, int lane) {
  int c = lane & 15, h = (lane >> 4) & 1;
  const unsigned short* p = base + (size_t)(n0 + c) * ld + k0 + 8 * h;
  v8u b0 = *reinterpret_cast<const v8u*>(p);
  v8u b1 = *reinterpret_cast<const v8u*>(p + 16);
  return __builtin_shufflevector(b0, b1, 0,1,2,3,4,5,6,7,8,9,10,11,12,13,14,15);
}

// Fused cross-wave LayerNorm over D=128 from sPre[16][132]; stores fp32 (+bf16).
DEV void ln_store(float (*sPre)[132], float (*sR1)[16], float (*sR2)[16],
                  float* sMu, float* sRs,
                  const float* g, const float* be,
                  float* outf, unsigned short* outbf,
                  size_t rowBase, int validRows) {
  __syncthreads();
  int r = threadIdx.x >> 4, seg = threadIdx.x & 15;
  float s1 = 0.f, s2 = 0.f;
#pragma unroll
  for (int e = 0; e < 8; ++e) { float x = sPre[r][seg * 8 + e]; s1 += x; s2 += x * x; }
  sR1[r][seg] = s1; sR2[r][seg] = s2;
  __syncthreads();
  if (threadIdx.x < 16) {
    int rr = threadIdx.x; float a = 0.f, b = 0.f;
#pragma unroll
    for (int e = 0; e < 16; ++e) { a += sR1[rr][e]; b += sR2[rr][e]; }
    float mu = a * (1.f / 128.f);
    float var = b * (1.f / 128.f) - mu * mu;
    sMu[rr] = mu; sRs[rr] = rsqrtf(var + 1e-5f);
  }
  __syncthreads();
  if (r < validRows) {
    float mu = sMu[r], rs = sRs[r];
    size_t rb = (rowBase + (size_t)r) * Dd;
#pragma unroll
    for (int e = 0; e < 8; ++e) {
      int d = seg * 8 + e;
      float o = (sPre[r][d] - mu) * rs * g[d] + be[d];
      outf[rb + d] = o;
      if (outbf) outbf[rb + d] = f2bf(o);
    }
  }
}

// ---------------- kernel 0: weight prep (transpose -> bf16) ----------------
__global__ void __launch_bounds__(256)
k_prep(const float* Wq, const float* Wk, const float* Wv,
       const float* W1, const float* W2,
       unsigned short* WqT, unsigned short* WkT, unsigned short* WvT,
       unsigned short* W1T, unsigned short* W2T) {
  int t = blockIdx.x * 256 + threadIdx.x;
  if (t < Dd * Dd) {                       // Wq/Wk/Wv [k][n] -> [n][k]
    int k = t >> 7, n = t & 127;
    WqT[n * Dd + k] = f2bf(Wq[t]);
    WkT[n * Dd + k] = f2bf(Wk[t]);
    WvT[n * Dd + k] = f2bf(Wv[t]);
  } else if (t < Dd * Dd + Dd * Ff) {      // W1 [k=128][f=512] -> [f][k]
    int u = t - Dd * Dd;
    int k = u / Ff, f = u % Ff;
    W1T[f * Dd + k] = f2bf(W1[u]);
  } else if (t < Dd * Dd + 2 * Dd * Ff) {  // W2 [f=512][d=128] -> [d][f]
    int u = t - Dd * Dd - Dd * Ff;
    int f = u / Dd, d = u % Dd;
    W2T[d * Ff + f] = f2bf(W2[u]);
  }
}

// ---------------- kernel 1: QKV + elementwise, producing sigq / ekT / ekvT --
__global__ void __launch_bounds__(256)
k_qkv(const float* __restrict__ data,
      const unsigned short* __restrict__ WkT,
      const unsigned short* __restrict__ WvT,
      const unsigned short* __restrict__ WqT,
      float* kq,                 // phase0: k (fp32), phase2: sigmoid(q)
      unsigned short* ekT, unsigned short* ekvT) {
  int wave = threadIdx.x >> 5, lane = threadIdx.x & 31;
  int r0 = blockIdx.x * 128 + wave * 16;          // flat row base (< 32000)
  const float* arow = data + (size_t)r0 * Dd;
  int cl = lane & 15, hh = (lane >> 4) & 1;

#pragma unroll 1
  for (int ph = 0; ph < 3; ++ph) {                // 0:k 1:v 2:q
    const unsigned short* WT = (ph == 0) ? WkT : (ph == 1 ? WvT : WqT);
    v8f acc[8] = {};
#pragma unroll
    for (int s = 0; s < 4; ++s) {
      v16u a = loadA_f32(arow, Dd, 32 * s, lane);
#pragma unroll
      for (int c = 0; c < 8; ++c)
        acc[c] = wmma_bf16(a, loadB_bf(WT, Dd, 16 * c, 32 * s, lane), acc[c]);
    }
#pragma unroll
    for (int c = 0; c < 8; ++c) {
#pragma unroll
      for (int m = 0; m < 8; ++m) {
        int row = r0 + m + 8 * hh;
        int d = 16 * c + cl;
        float v = acc[c][m];
        size_t idx = (size_t)row * Dd + d;
        if (ph == 0) {
          kq[idx] = v;                                     // raw k
        } else if (ph == 1) {
          float e = expf(kq[idx]);                         // ek
          int bb = row / Nn, n = row - bb * Nn;
          size_t t = ((size_t)(bb * Dd + d)) * NP + n;     // transposed, K-padded
          ekT[t] = f2bf(e);
          ekvT[t] = f2bf(e * v);                           // ek * v
        } else {
          kq[idx] = 1.f / (1.f + expf(-v));                // sigmoid(q), reuse buf
        }
      }
    }
  }
}

// ------- kernel 2: AFT num/den GEMM + sigq*num/den + residual + LN1 --------
__global__ void __launch_bounds__(256)
k_aft_ln1(const float* __restrict__ data,
          const float* __restrict__ dist, const float* __restrict__ mask,
          const float* __restrict__ alpha,
          const float* __restrict__ sigq,
          const unsigned short* __restrict__ ekT,
          const unsigned short* __restrict__ ekvT,
          const float* __restrict__ g1, const float* __restrict__ be1,
          float* out1f, unsigned short* out1bf) {
  __shared__ alignas(16) unsigned short sA[2][16 * 40];    // double-buffered A tile
  __shared__ float sPre[16][132];
  __shared__ float sR1[16][16], sR2[16][16];
  __shared__ float sMu[16], sRs[16];

  int b = blockIdx.y;
  int i0 = blockIdx.x * 16;                                // row strip in batch
  int wave = threadIdx.x >> 5, lane = threadIdx.x & 31;
  float al = alpha[0];
  const float* distB = dist + (size_t)b * Nn * Nn;
  const float* maskB = mask + (size_t)b * Nn * Nn;
  const unsigned short* ekTb  = ekT  + (size_t)b * Dd * NP;
  const unsigned short* ekvTb = ekvT + (size_t)b * Dd * NP;

  v8f accN = {}, accD = {};
#pragma unroll 1
  for (int s = 0; s < 32; ++s) {                           // K = 1024 (padded)
    unsigned short* buf = sA[s & 1];
    // stage exp(alpha*dist + mask) tile [16 x 32] as bf16 (zero outside bounds)
#pragma unroll
    for (int e = threadIdx.x; e < 512; e += 256) {
      int r = e >> 5, j = e & 31;
      int i = i0 + r, jj = 32 * s + j;
      float w = 0.f;
      if (i < Nn && jj < Nn) {
        size_t ix = (size_t)i * Nn + jj;
        w = expf(al * distB[ix] + maskB[ix]);
        if (jj + 32 < Nn) {                                // prefetch next k-tile
          __builtin_prefetch(&distB[ix + 32], 0, 1);
          __builtin_prefetch(&maskB[ix + 32], 0, 1);
        }
      }
      buf[r * 40 + j] = f2bf(w);
    }
    __syncthreads();
    v16u a  = loadA_bf(buf, 40, 0, lane);                  // ds_load_b128 x2
    v16u bn = loadB_bf(ekvTb, NP, 16 * wave, 32 * s, lane);
    v16u bd = loadB_bf(ekTb,  NP, 16 * wave, 32 * s, lane);
    accN = wmma_bf16(a, bn, accN);
    accD = wmma_bf16(a, bd, accD);
  }

  // epilogue: aafm = sigq * num/den ; pre = data + aafm -> LDS
  int cl = lane & 15, hh = (lane >> 4) & 1;
#pragma unroll
  for (int m = 0; m < 8; ++m) {
    int r = m + 8 * hh;
    int i = i0 + r;
    int ii = (i < Nn) ? i : (Nn - 1);                      // clamp loads for pad rows
    int d = 16 * wave + cl;
    size_t idx = ((size_t)(b * Nn + ii)) * Dd + d;
    float pre = data[idx] + sigq[idx] * accN[m] / accD[m];
    sPre[r][d] = pre;
  }
  ln_store(sPre, sR1, sR2, sMu, sRs, g1, be1, out1f, out1bf,
           (size_t)b * Nn + i0, Nn - i0);
}

// ---------------- kernel 3: FFN1 = relu(out1 @ W1 + b1) -> bf16 ------------
__global__ void __launch_bounds__(256)
k_ffn1(const unsigned short* __restrict__ out1bf,
       const unsigned short* __restrict__ W1T,
       const float* __restrict__ b1, unsigned short* hbf) {
  int wave = threadIdx.x >> 5, lane = threadIdx.x & 31;
  int r0 = blockIdx.x * 128 + wave * 16;
  const unsigned short* arow = out1bf + (size_t)r0 * Dd;
  int cl = lane & 15, hh = (lane >> 4) & 1;

#pragma unroll 1
  for (int cg = 0; cg < 4; ++cg) {                         // 4 col-groups of 128
    v8f acc[8] = {};
#pragma unroll
    for (int s = 0; s < 4; ++s) {
      v16u a = loadA_bf(arow, Dd, 32 * s, lane);
#pragma unroll
      for (int c = 0; c < 8; ++c)
        acc[c] = wmma_bf16(a, loadB_bf(W1T, Dd, cg * 128 + 16 * c, 32 * s, lane), acc[c]);
    }
#pragma unroll
    for (int c = 0; c < 8; ++c) {
#pragma unroll
      for (int m = 0; m < 8; ++m) {
        int row = r0 + m + 8 * hh;
        int f = cg * 128 + 16 * c + cl;
        float v = acc[c][m] + b1[f];
        hbf[(size_t)row * Ff + f] = f2bf(v > 0.f ? v : 0.f);
      }
    }
  }
}

// -------- kernel 4: FFN2 = h @ W2 + b2, + residual(out1) + LN2 -> d_out ----
__global__ void __launch_bounds__(256)
k_ffn2_ln2(const unsigned short* __restrict__ hbf,
           const unsigned short* __restrict__ W2T,
           const float* __restrict__ b2,
           const float* __restrict__ g2, const float* __restrict__ be2,
           float* io /* holds out1 fp32, overwritten with final output */) {
  __shared__ float sPre[16][132];
  __shared__ float sR1[16][16], sR2[16][16];
  __shared__ float sMu[16], sRs[16];

  int wave = threadIdx.x >> 5, lane = threadIdx.x & 31;
  int r0 = blockIdx.x * 16;
  const unsigned short* arow = hbf + (size_t)r0 * Ff;

  v8f acc = {};
#pragma unroll 1
  for (int s = 0; s < 16; ++s) {                           // K = 512
    v16u a = loadA_bf(arow, Ff, 32 * s, lane);
    v16u bb = loadB_bf(W2T, Ff, 16 * wave, 32 * s, lane);
    acc = wmma_bf16(a, bb, acc);
  }

  int cl = lane & 15, hh = (lane >> 4) & 1;
#pragma unroll
  for (int m = 0; m < 8; ++m) {
    int r = m + 8 * hh;
    int d = 16 * wave + cl;
    size_t idx = (size_t)(r0 + r) * Dd + d;
    sPre[r][d] = io[idx] + acc[m] + b2[d];                 // residual + bias
  }
  ln_store(sPre, sR1, sR2, sMu, sRs, g2, be2, io, nullptr, (size_t)r0, 16);
}

// ---------------- host launcher ----------------
extern "C" void kernel_launch(void* const* d_in, const int* in_sizes, int n_in,
                              void* d_out, int out_size, void* d_ws, size_t ws_size,
                              hipStream_t stream) {
  (void)in_sizes; (void)n_in; (void)out_size; (void)ws_size;
  const float* data  = (const float*)d_in[0];
  const float* dist  = (const float*)d_in[1];
  const float* mask  = (const float*)d_in[2];
  const float* Wq    = (const float*)d_in[3];
  const float* Wk    = (const float*)d_in[4];
  const float* Wv    = (const float*)d_in[5];
  const float* alpha = (const float*)d_in[6];
  const float* g1    = (const float*)d_in[7];
  const float* be1   = (const float*)d_in[8];
  const float* g2    = (const float*)d_in[9];
  const float* be2   = (const float*)d_in[10];
  const float* W1    = (const float*)d_in[11];
  const float* b1    = (const float*)d_in[12];
  const float* W2    = (const float*)d_in[13];
  const float* b2    = (const float*)d_in[14];
  float* out = (float*)d_out;                     // also used as out1 fp32 scratch

  char* w = (char*)d_ws;
  auto take = [&](size_t bytes) -> void* {
    void* p = (void*)w;
    w += (bytes + 255) & ~(size_t)255;
    return p;
  };
  unsigned short* WqT  = (unsigned short*)take((size_t)Dd * Dd * 2);
  unsigned short* WkT  = (unsigned short*)take((size_t)Dd * Dd * 2);
  unsigned short* WvT  = (unsigned short*)take((size_t)Dd * Dd * 2);
  unsigned short* W1T  = (unsigned short*)take((size_t)Ff * Dd * 2);
  unsigned short* W2T  = (unsigned short*)take((size_t)Dd * Ff * 2);
  float*          kq   = (float*)take((size_t)BN * Dd * 4);         // k, then sigq
  unsigned short* ekT  = (unsigned short*)take((size_t)Bd * Dd * NP * 2);
  unsigned short* ekvT = (unsigned short*)take((size_t)Bd * Dd * NP * 2);
  unsigned short* o1bf = (unsigned short*)take((size_t)BN * Dd * 2);
  unsigned short* hbf  = (unsigned short*)take((size_t)BN * Ff * 2);

  k_prep<<<dim3(704), dim3(256), 0, stream>>>(Wq, Wk, Wv, W1, W2,
                                              WqT, WkT, WvT, W1T, W2T);
  k_qkv<<<dim3(BN / 128), dim3(256), 0, stream>>>(data, WkT, WvT, WqT,
                                                  kq, ekT, ekvT);
  k_aft_ln1<<<dim3(63, Bd), dim3(256), 0, stream>>>(data, dist, mask, alpha,
                                                    kq, ekT, ekvT, g1, be1,
                                                    out, o1bf);
  k_ffn1<<<dim3(BN / 128), dim3(256), 0, stream>>>(o1bf, W1T, b1, hbf);
  k_ffn2_ln2<<<dim3(BN / 16), dim3(256), 0, stream>>>(hbf, W2T, b2, g2, be2, out);
}
